// OTLoss_12154757448399
// MI455X (gfx1250) — matrix operand order repaired and checked by reading, not compiled, not run on hardware
//
#include <hip/hip_runtime.h>
#include <hip/hip_bf16.h>
#include <stdint.h>

typedef __attribute__((ext_vector_type(16))) __bf16 v16bf;
typedef __attribute__((ext_vector_type(8)))  float  v8f;
typedef __attribute__((ext_vector_type(4)))  unsigned int u32x4;
typedef __attribute__((ext_vector_type(8)))  int i32x8;
typedef __attribute__((ext_vector_type(4)))  int i32x4;

#define B_    8
#define S1_   2048
#define S2_   2048
#define DIN_  768
#define DOUT_ 1600
#define REG_  0.1f
#define SINK_ITERS_ 100

#if defined(__has_builtin)
#  if __has_builtin(__builtin_amdgcn_tensor_load_to_lds) && \
      __has_builtin(__builtin_amdgcn_s_wait_tensorcnt)
#    define HAVE_TDM 1
#  endif
#endif
#ifndef HAVE_TDM
#  define HAVE_TDM 0
#endif

#if HAVE_TDM
// ---------------------------------------------------------------------------
// Tensor Data Mover: 2-D bf16 tile load, tile_w == 32 elements (64B rows),
// LDS destination padded to 40-element rows (pad_interval=16DW, pad_amount=4DW)
// D# per CDNA5 ISA.  6-arg builtin form (clang-23).
// ---------------------------------------------------------------------------
__device__ __forceinline__ void tdm_load_tile32_bf16(
    unsigned lds_byte_addr, const __bf16* gsrc,
    unsigned tensor_w, unsigned tensor_h, unsigned long long row_stride,
    unsigned tile_h) {
    unsigned long long ga = (unsigned long long)(uintptr_t)gsrc;

    u32x4 g0;
    g0[0] = 1u;                                   // count=1 (valid), user mode
    g0[1] = lds_byte_addr;                        // lds_addr [63:32]
    g0[2] = (unsigned)(ga & 0xFFFFFFFFull);       // global_addr[31:0]
    g0[3] = (unsigned)((ga >> 32) & 0x1FFFFFFull) // global_addr[56:32]
          | (2u << 30);                           // type=2 ("image")

    // group1: data_size=2B, pad_enable, pad_interval=16DW, pad_amount=4DW
    unsigned w0 = (1u << 16) | (1u << 20) | (3u << 22) | (3u << 25);
    unsigned long long q0 = (unsigned long long)w0
                          | ((unsigned long long)(tensor_w & 0xFFFFu) << 48);
    unsigned long long q1 = (unsigned long long)(tensor_w >> 16)
                          | ((unsigned long long)tensor_h << 16)
                          | ((unsigned long long)32u << 48);       // tile_dim0=32
    unsigned long long q2 = (unsigned long long)tile_h             // tile_dim1
                          | ((row_stride & 0xFFFFFFFFull) << 32);  // dim0_stride lo
    unsigned long long q3 = (row_stride >> 32) & 0xFFFFull;        // dim0_stride hi

    union { unsigned long long q[4]; i32x8 v; } g1u;
    g1u.q[0] = q0; g1u.q[1] = q1; g1u.q[2] = q2; g1u.q[3] = q3;

    i32x4 g2 = {};   // 2-D tensor: dims 2+ unused
    i32x4 g3 = {};
    i32x8 g4 = {};   // extra operand of the 6-arg builtin form: unused, zero
    __builtin_amdgcn_tensor_load_to_lds(g0, g1u.v, g2, g3, g4, 0);
}
#endif

// ---------------------------------------------------------------------------
// init: zero loss + student sum-of-squares, set u,v to uniform marginals
// ---------------------------------------------------------------------------
__global__ void init_kernel(float* __restrict__ out, float* __restrict__ ns2,
                            float* __restrict__ u, float* __restrict__ v) {
    int t = blockIdx.x * blockDim.x + threadIdx.x;
    if (t == 0) out[0] = 0.f;
    if (t < B_ * S2_) ns2[t] = 0.f;
    if (t < B_ * S1_) u[t] = 1.f / S1_;
    if (t < B_ * S2_) v[t] = 1.f / S2_;
}

// ---------------------------------------------------------------------------
// teacher prep: per-row L2 normalize, write bf16
// ---------------------------------------------------------------------------
__global__ __launch_bounds__(256) void teacher_prep_kernel(
    const float* __restrict__ t_in, __bf16* __restrict__ tn) {
    __shared__ float red[256];
    const int row = blockIdx.x;
    const float* src = t_in + (size_t)row * DOUT_;
    float ss = 0.f;
    for (int k = threadIdx.x; k < DOUT_; k += 256) { float x = src[k]; ss += x * x; }
    red[threadIdx.x] = ss;
    __syncthreads();
    for (int s = 128; s > 0; s >>= 1) {
        if (threadIdx.x < s) red[threadIdx.x] += red[threadIdx.x + s];
        __syncthreads();
    }
    const float rn = 1.f / fmaxf(sqrtf(red[0]), 1e-12f);
    __bf16* dst = tn + (size_t)row * DOUT_;
    for (int k = threadIdx.x; k < DOUT_; k += 256) dst[k] = (__bf16)(src[k] * rn);
}

// ---------------------------------------------------------------------------
// shared fragment builders (ISA §7.12.2 layouts), LDS rows padded to 40 elems
// ---------------------------------------------------------------------------
__device__ __forceinline__ void build_a_frag(const __bf16 (*As)[40], int mbase,
                                             int l16, int hl, v16bf& a) {
#pragma unroll
    for (int r = 0; r < 8; ++r) {
        int kb = (r < 4 ? 2 * r : 16 + 2 * (r - 4)) + hl * 8;
        a[2 * r]     = As[mbase + l16][kb];
        a[2 * r + 1] = As[mbase + l16][kb + 1];
    }
}
__device__ __forceinline__ void build_b_frag(const __bf16 (*Bs)[40], int nbase,
                                             int l16, int hl, v16bf& b) {
#pragma unroll
    for (int r = 0; r < 8; ++r) {
        int kb = (r < 4 ? 2 * r : 16 + 2 * (r - 4)) + hl * 8;
        b[2 * r]     = Bs[nbase + l16][kb];
        b[2 * r + 1] = Bs[nbase + l16][kb + 1];
    }
}

// ---------------------------------------------------------------------------
// GEMM1: s = student(16384x768) @ W(768x1600) + b  -> bf16, plus row sumsq
// workgroup tile 128x64, 8 waves of 32x32 (2x2 WMMA 16x16x32 bf16)
// LDS-staged coalesced bf16 writeback.
// ---------------------------------------------------------------------------
__global__ __launch_bounds__(256) void linear_gemm_kernel(
    const float* __restrict__ S, const float* __restrict__ W,
    const float* __restrict__ bias, __bf16* __restrict__ sOut,
    float* __restrict__ ns2) {
    __shared__ __bf16 As[128][40];
    __shared__ __bf16 Bs[64][40];
    __shared__ __bf16 Co[128][72];
    __shared__ float  rowss[128];

    const int m0 = blockIdx.x * 128;
    const int n0 = blockIdx.y * 64;
    const int tid = threadIdx.x;
    if (tid < 128) rowss[tid] = 0.f;

    const int wid = tid >> 5, lane = tid & 31;
    const int wy = wid >> 1, wx = wid & 1;
    const int hl = lane >> 4, l16 = lane & 15;

    v8f acc[2][2] = {};

    for (int k0 = 0; k0 < DIN_; k0 += 32) {
        __syncthreads();
        {   // A tile 128x32 fp32 -> bf16 LDS
            int r = tid >> 1, c = (tid & 1) * 16;
            const float* p = S + (size_t)(m0 + r) * DIN_ + k0 + c;
#pragma unroll
            for (int i = 0; i < 16; i += 4) {
                float4 f = *(const float4*)(p + i);
                As[r][c + i + 0] = (__bf16)f.x; As[r][c + i + 1] = (__bf16)f.y;
                As[r][c + i + 2] = (__bf16)f.z; As[r][c + i + 3] = (__bf16)f.w;
            }
        }
        {   // B tile W[k0..+32)[n0..+64) -> Bs[n][k]
            int kk = tid >> 3, nn = (tid & 7) * 8;
            const float* p = W + (size_t)(k0 + kk) * DOUT_ + n0 + nn;
#pragma unroll
            for (int i = 0; i < 8; i += 4) {
                float4 f = *(const float4*)(p + i);
                Bs[nn + i + 0][kk] = (__bf16)f.x; Bs[nn + i + 1][kk] = (__bf16)f.y;
                Bs[nn + i + 2][kk] = (__bf16)f.z; Bs[nn + i + 3][kk] = (__bf16)f.w;
            }
        }
        __syncthreads();

        v16bf afr[2], bfr[2];
        build_a_frag(As, wy * 32, l16, hl, afr[0]);
        build_a_frag(As, wy * 32 + 16, l16, hl, afr[1]);
        build_b_frag(Bs, wx * 32, l16, hl, bfr[0]);
        build_b_frag(Bs, wx * 32 + 16, l16, hl, bfr[1]);
#pragma unroll
        for (int tm = 0; tm < 2; ++tm)
#pragma unroll
            for (int tn = 0; tn < 2; ++tn)
                acc[tm][tn] = __builtin_amdgcn_wmma_f32_16x16x32_bf16(
                    false, afr[tm], false, bfr[tn], (short)0, acc[tm][tn], false, false);
    }
    __syncthreads();

    // epilogue: +bias, stage bf16 tile in LDS, accumulate row sum-of-squares
#pragma unroll
    for (int tn = 0; tn < 2; ++tn) {
        int nloc = wx * 32 + tn * 16 + l16;
        float bb = bias[n0 + nloc];
#pragma unroll
        for (int tm = 0; tm < 2; ++tm) {
#pragma unroll
            for (int r = 0; r < 8; ++r) {
                int mloc = wy * 32 + tm * 16 + r + hl * 8;
                float val = acc[tm][tn][r] + bb;
                Co[mloc][nloc] = (__bf16)val;
                atomicAdd(&rowss[mloc], val * val);
            }
        }
    }
    __syncthreads();
    {   // coalesced writeback: thread -> 32 contiguous bf16 (64B)
        int row = tid >> 1, coff = (tid & 1) * 32;
        __bf16* dst = sOut + (size_t)(m0 + row) * DOUT_ + n0 + coff;
        const __bf16* srcp = &Co[row][coff];
#pragma unroll
        for (int i = 0; i < 32; i += 8)
            *(uint4*)(dst + i) = *(const uint4*)(srcp + i);
    }
    if (tid < 128) atomicAdd(&ns2[m0 + tid], rowss[tid]);
}

// ---------------------------------------------------------------------------
// ns2 -> reciprocal norm
// ---------------------------------------------------------------------------
__global__ void rns_kernel(float* __restrict__ ns2) {
    int t = blockIdx.x * blockDim.x + threadIdx.x;
    if (t < B_ * S2_) ns2[t] = 1.f / fmaxf(sqrtf(ns2[t]), 1e-12f);
}

// ---------------------------------------------------------------------------
// GEMM2: per batch, Kmat[i,j] = exp(-5*(1 - tn_i . s_j / ||s_j||))
// TDM double-buffered tile loads (DMA for step k+1 overlaps WMMA of step k),
// LDS-staged coalesced f32 writeback.
// ---------------------------------------------------------------------------
__global__ __launch_bounds__(256) void cost_gemm_kernel(
    const __bf16* __restrict__ tn, const __bf16* __restrict__ sn,
    const float* __restrict__ rns, float* __restrict__ Kmat) {
    __shared__ __bf16 As[2][128][40];
    __shared__ __bf16 Bs[2][64][40];
    __shared__ float  Cs[128][68];

    const int b = blockIdx.z;
    const int m0 = blockIdx.x * 128, n0 = blockIdx.y * 64;
    const __bf16* A  = tn + (size_t)b * S1_ * DOUT_;
    const __bf16* Bm = sn + (size_t)b * S2_ * DOUT_;

    const int tid = threadIdx.x, wid = tid >> 5, lane = tid & 31;
    const int wy = wid >> 1, wx = wid & 1;
    const int hl = lane >> 4, l16 = lane & 15;

    v8f acc[2][2] = {};

#if HAVE_TDM
    const bool issuer = (wid == 0);
    const unsigned ldsA[2] = { (unsigned)(uintptr_t)&As[0][0][0],
                               (unsigned)(uintptr_t)&As[1][0][0] };
    const unsigned ldsB[2] = { (unsigned)(uintptr_t)&Bs[0][0][0],
                               (unsigned)(uintptr_t)&Bs[1][0][0] };
    if (issuer) {   // prologue: DMA first tiles into buffer 0
        tdm_load_tile32_bf16(ldsA[0], A + (size_t)m0 * DOUT_, DOUT_, S1_, DOUT_, 128);
        tdm_load_tile32_bf16(ldsB[0], Bm + (size_t)n0 * DOUT_, DOUT_, S2_, DOUT_, 64);
    }
    int buf = 0;
    for (int k0 = 0; k0 < DOUT_; k0 += 32) {
        if (issuer) __builtin_amdgcn_s_wait_tensorcnt(0);
        __syncthreads();                    // current buffer ready for all waves
        if (issuer && (k0 + 32 < DOUT_)) {  // overlap: DMA next tiles
            tdm_load_tile32_bf16(ldsA[buf ^ 1], A + (size_t)m0 * DOUT_ + k0 + 32,
                                 DOUT_, S1_, DOUT_, 128);
            tdm_load_tile32_bf16(ldsB[buf ^ 1], Bm + (size_t)n0 * DOUT_ + k0 + 32,
                                 DOUT_, S2_, DOUT_, 64);
        }
        v16bf afr[2], bfr[2];
        build_a_frag(As[buf], wy * 32, l16, hl, afr[0]);
        build_a_frag(As[buf], wy * 32 + 16, l16, hl, afr[1]);
        build_b_frag(Bs[buf], wx * 32, l16, hl, bfr[0]);
        build_b_frag(Bs[buf], wx * 32 + 16, l16, hl, bfr[1]);
#pragma unroll
        for (int tm = 0; tm < 2; ++tm)
#pragma unroll
            for (int tn2 = 0; tn2 < 2; ++tn2)
                acc[tm][tn2] = __builtin_amdgcn_wmma_f32_16x16x32_bf16(
                    false, afr[tm], false, bfr[tn2], (short)0, acc[tm][tn2], false, false);
        __syncthreads();                    // all waves done with buf before reuse
        buf ^= 1;
    }
#else
    for (int k0 = 0; k0 < DOUT_; k0 += 32) {
        __syncthreads();
        {   int r = tid >> 1, c = (tid & 1) * 16;
            const __bf16* p = A + (size_t)(m0 + r) * DOUT_ + k0 + c;
            *(uint4*)&As[0][r][c]     = *(const uint4*)p;
            *(uint4*)&As[0][r][c + 8] = *(const uint4*)(p + 8);
        }
        {   int n = tid >> 2, ko = (tid & 3) * 8;
            const __bf16* p = Bm + (size_t)(n0 + n) * DOUT_ + k0 + ko;
            *(uint4*)&Bs[0][n][ko] = *(const uint4*)p;
        }
        __syncthreads();
        v16bf afr[2], bfr[2];
        build_a_frag(As[0], wy * 32, l16, hl, afr[0]);
        build_a_frag(As[0], wy * 32 + 16, l16, hl, afr[1]);
        build_b_frag(Bs[0], wx * 32, l16, hl, bfr[0]);
        build_b_frag(Bs[0], wx * 32 + 16, l16, hl, bfr[1]);
#pragma unroll
        for (int tm = 0; tm < 2; ++tm)
#pragma unroll
            for (int tn2 = 0; tn2 < 2; ++tn2)
                acc[tm][tn2] = __builtin_amdgcn_wmma_f32_16x16x32_bf16(
                    false, afr[tm], false, bfr[tn2], (short)0, acc[tm][tn2], false, false);
    }
#endif

    // epilogue: sim -> exp(-5(1-sim)), stage in LDS, coalesced b128 writeback
#pragma unroll
    for (int tn2 = 0; tn2 < 2; ++tn2) {
        int nloc = wx * 32 + tn2 * 16 + l16;
        float rn = rns[b * S2_ + n0 + nloc];
#pragma unroll
        for (int tm = 0; tm < 2; ++tm) {
#pragma unroll
            for (int r = 0; r < 8; ++r) {
                int mloc = wy * 32 + tm * 16 + r + hl * 8;
                float sim = acc[tm][tn2][r] * rn;
                Cs[mloc][nloc] = __expf(-5.f * (1.f - sim));
            }
        }
    }
    __syncthreads();
    {   // thread -> 2 rows x 16 contiguous floats (4x b128 each)
#pragma unroll
        for (int rr = 0; rr < 2; ++rr) {
            int row = (tid >> 2) + rr * 64;
            int coff = (tid & 3) * 16;
            float* dst = Kmat + ((size_t)b * S1_ + m0 + row) * S2_ + n0 + coff;
            const float* srcp = &Cs[row][coff];
#pragma unroll
            for (int i = 0; i < 16; i += 4)
                *(float4*)(dst + i) = *(const float4*)(srcp + i);
        }
    }
}

// ---------------------------------------------------------------------------
// Sinkhorn v-update: v_j = (1/m) / (K^T u)_j   (block = 32 cols x 8 row-groups)
// ---------------------------------------------------------------------------
__global__ __launch_bounds__(256) void sink_v_kernel(
    const float* __restrict__ Kmat, const float* __restrict__ u,
    float* __restrict__ v) {
    __shared__ float red[8][32];
    const int b = blockIdx.y;
    const int lane = threadIdx.x & 31;
    const int g = threadIdx.x >> 5;
    const int j = blockIdx.x * 32 + lane;
    const float* Kb = Kmat + (size_t)b * S1_ * S2_;
    const float* ub = u + b * S1_;
    float acc = 0.f;
    for (int i = g; i < S1_; i += 8) {
        __builtin_prefetch(&Kb[(size_t)(i + 32) * S2_ + j], 0, 1);
        acc += Kb[(size_t)i * S2_ + j] * ub[i];
    }
    red[g][lane] = acc;
    __syncthreads();
    if (g == 0) {
        float s = red[0][lane] + red[1][lane] + red[2][lane] + red[3][lane]
                + red[4][lane] + red[5][lane] + red[6][lane] + red[7][lane];
        v[b * S2_ + j] = (1.f / S2_) / s;
    }
}

// ---------------------------------------------------------------------------
// Sinkhorn u-update: u_i = (1/n) / (K v)_i ; one wave32 per row, float4 stream
// ---------------------------------------------------------------------------
__global__ __launch_bounds__(256) void sink_u_kernel(
    const float* __restrict__ Kmat, const float* __restrict__ v,
    float* __restrict__ u) {
    const int b = blockIdx.y;
    const int g = threadIdx.x >> 5, lane = threadIdx.x & 31;
    const int i = blockIdx.x * 8 + g;
    const float* Kb = Kmat + ((size_t)b * S1_ + i) * S2_;
    const float* vb = v + b * S2_;
    float acc = 0.f;
    for (int j = lane * 4; j < S2_; j += 128) {
        __builtin_prefetch(&Kb[j + 1024], 0, 1);
        float4 kv = *(const float4*)(Kb + j);
        float4 vv = *(const float4*)(vb + j);
        acc += kv.x * vv.x + kv.y * vv.y + kv.z * vv.z + kv.w * vv.w;
    }
#pragma unroll
    for (int off = 16; off > 0; off >>= 1)
        acc += __shfl_down(acc, off, 32);
    if (lane == 0) u[b * S1_ + i] = (1.f / S1_) / acc;
}

// ---------------------------------------------------------------------------
// loss = sum_{b,i,j} u_i K_ij v_j * (-reg * ln K_ij) / B
// ---------------------------------------------------------------------------
__global__ __launch_bounds__(256) void loss_kernel(
    const float* __restrict__ Kmat, const float* __restrict__ u,
    const float* __restrict__ v, float* __restrict__ out) {
    __shared__ float red[256];
    const int b = blockIdx.y, i = blockIdx.x;
    const float* Kb = Kmat + ((size_t)b * S1_ + i) * S2_;
    const float* vb = v + b * S2_;
    const float ui = u[b * S1_ + i];
    float acc = 0.f;
    for (int j = threadIdx.x * 4; j < S2_; j += 1024) {
        float4 kv = *(const float4*)(Kb + j);
        float4 vv = *(const float4*)(vb + j);
        acc += kv.x * vv.x * __logf(kv.x) + kv.y * vv.y * __logf(kv.y)
             + kv.z * vv.z * __logf(kv.z) + kv.w * vv.w * __logf(kv.w);
    }
    acc *= -REG_ * ui;
    red[threadIdx.x] = acc;
    __syncthreads();
    for (int s = 128; s > 0; s >>= 1) {
        if (threadIdx.x < s) red[threadIdx.x] += red[threadIdx.x + s];
        __syncthreads();
    }
    if (threadIdx.x == 0) atomicAdd(out, red[0] * (1.f / B_));
}

// ---------------------------------------------------------------------------
extern "C" void kernel_launch(void* const* d_in, const int* in_sizes, int n_in,
                              void* d_out, int out_size, void* d_ws, size_t ws_size,
                              hipStream_t stream) {
    const float* teacher = (const float*)d_in[0];   // (B,S1,DOUT) fp32
    const float* student = (const float*)d_in[1];   // (B,S2,DIN)  fp32
    const float* W       = (const float*)d_in[2];   // (DIN,DOUT)  fp32
    const float* bias    = (const float*)d_in[3];   // (DOUT,)     fp32
    float* out = (float*)d_out;

    char* ws = (char*)d_ws;
    size_t off = 0;
    auto take = [&](size_t bytes) -> char* {
        char* p = ws + off;
        off = (off + bytes + 255) & ~(size_t)255;
        return p;
    };
    __bf16* s_bf  = (__bf16*)take((size_t)B_ * S2_ * DOUT_ * 2); // 52.4 MB
    __bf16* tn_bf = (__bf16*)take((size_t)B_ * S1_ * DOUT_ * 2); // 52.4 MB
    float*  Kmat  = (float*)take((size_t)B_ * S1_ * S2_ * 4);    // 134.2 MB (L2 resident)
    float*  rns   = (float*)take((size_t)B_ * S2_ * 4);
    float*  u     = (float*)take((size_t)B_ * S1_ * 4);
    float*  v     = (float*)take((size_t)B_ * S2_ * 4);

    init_kernel<<<(B_ * S2_ + 255) / 256, 256, 0, stream>>>(out, rns, u, v);
    teacher_prep_kernel<<<B_ * S1_, 256, 0, stream>>>(teacher, tn_bf);
    linear_gemm_kernel<<<dim3((B_ * S2_) / 128, DOUT_ / 64), 256, 0, stream>>>(
        student, W, bias, s_bf, rns);
    rns_kernel<<<(B_ * S2_ + 255) / 256, 256, 0, stream>>>(rns);
    cost_gemm_kernel<<<dim3(S1_ / 128, S2_ / 64, B_), 256, 0, stream>>>(
        tn_bf, s_bf, rns, Kmat);
    for (int it = 0; it < SINK_ITERS_; ++it) {
        sink_v_kernel<<<dim3(S2_ / 32, B_), 256, 0, stream>>>(Kmat, u, v);
        sink_u_kernel<<<dim3(S1_ / 8, B_), 256, 0, stream>>>(Kmat, v, u);
    }
    loss_kernel<<<dim3(S1_, B_), 256, 0, stream>>>(Kmat, u, v, out);
}
